// GLMRNN_11922829214442
// MI455X (gfx1250) — matrix operand order repaired and checked by reading, not compiled, not run on hardware
//
#include <hip/hip_runtime.h>
#include <hip/hip_bf16.h>

#define NN   2048
#define TT   4096
#define NT   (NN * TT)
#define NBLK 128
#define RPB  16          // rows per block (one WMMA M-tile)
#define NTHR 256         // 8 waves
#define NWAV 8
#define DTCONST 0.1f

typedef __attribute__((ext_vector_type(16))) __bf16 v16bf;
typedef __attribute__((ext_vector_type(8)))  __bf16 v8bf;
typedef __attribute__((ext_vector_type(8)))  float  v8f;

// ---------------- deterministic counter-based RNG + Poisson ----------------
__device__ __forceinline__ unsigned pcg_next(unsigned &state) {
  state = state * 747796405u + 2891336453u;
  unsigned w = ((state >> ((state >> 28u) + 4u)) ^ state) * 277803737u;
  return (w >> 22u) ^ w;
}
__device__ __forceinline__ float u01(unsigned r) {
  return (float)(r >> 8) * (1.0f / 16777216.0f) + 1.0e-8f;
}
__device__ __forceinline__ float poisson_sample(unsigned &st, float lam) {
  lam = fminf(fmaxf(lam, 0.0f), 1.0e6f);
  if (lam < 12.0f) {             // Knuth inversion for small rates
    float L = __expf(-lam);
    float p = 1.0f;
    int k = -1;
    do { k++; p *= u01(pcg_next(st)); } while (p > L && k < 64);
    return (float)k;
  }
  // normal approximation for large rates (Box-Muller)
  float u1 = u01(pcg_next(st));
  float u2 = u01(pcg_next(st));
  float z  = __fsqrt_rn(-2.0f * __logf(u1)) * __cosf(6.2831853f * u2);
  return floorf(fmaxf(0.0f, lam + __fsqrt_rn(lam) * z + 0.5f));
}

// ---------------- device-wide split barrier (L2 atomics) ----------------
__device__ __forceinline__ void grid_barrier(unsigned* cnt, unsigned* gen, unsigned nblk) {
  __syncthreads();
  if (threadIdx.x == 0) {
    __threadfence();   // publish this block's x-buffer writes device-wide
    unsigned g = __hip_atomic_load(gen, __ATOMIC_RELAXED, __HIP_MEMORY_SCOPE_AGENT);
    if (atomicAdd(cnt, 1u) == nblk - 1u) {
      __hip_atomic_store(cnt, 0u, __ATOMIC_RELAXED, __HIP_MEMORY_SCOPE_AGENT);
      __hip_atomic_store(gen, g + 1u, __ATOMIC_RELEASE, __HIP_MEMORY_SCOPE_AGENT);
    } else {
      while (__hip_atomic_load(gen, __ATOMIC_ACQUIRE, __HIP_MEMORY_SCOPE_AGENT) == g)
        __builtin_amdgcn_s_sleep(1);
    }
  }
  __syncthreads();
}

// ---------------- init: seed x broadcast buffer + reset barrier ----------------
__global__ void glmrnn_init(__bf16* __restrict__ xb0, unsigned* cnt, unsigned* gen) {
  int i = blockIdx.x * blockDim.x + threadIdx.x;
  if (i < NN) xb0[i] = (__bf16)1.0f;
  if (i == 0) { *cnt = 0u; *gen = 0u; }
}

// ---------------- persistent recurrence kernel ----------------
// Block b owns rows [b*16, b*16+16). W tile cached in LDS as bf16 (64 KB).
// Each of the 8 waves computes a 16x16x32 WMMA over its 256-wide K slice,
// with x replicated across the 16 B-columns; column 0 of D = partial dots.
__global__ __launch_bounds__(NTHR, 1)
void glmrnn_persistent(const float* __restrict__ W, const float* __restrict__ B,
                       const float* __restrict__ tau, float* __restrict__ out,
                       __bf16* __restrict__ xb0, __bf16* __restrict__ xb1,
                       unsigned* cnt, unsigned* gen)
{
  __shared__ __bf16 Wt[RPB * NN];        // 64 KB: this block's 16 rows of W, bf16
  __shared__ float  red[NWAV][RPB];      // cross-wave partial-sum reduce

  const int tid  = threadIdx.x;
  const int lane = tid & 31;
  const int wv   = tid >> 5;
  const int r0   = blockIdx.x * RPB;

  // One-time stage: global f32 W rows -> LDS bf16 tile (coalesced).
  for (int i = tid; i < RPB * NN; i += NTHR) {
    int rr = i >> 11;                    // i / 2048
    int k  = i & (NN - 1);
    Wt[i] = (__bf16)W[(size_t)(r0 + rr) * NN + k];
  }
  __syncthreads();

  const float decay = 1.0f - DTCONST / tau[0];

  // Per-row recurrent state lives in registers of threads 0..15.
  float xr = 1.0f, sr = 1.0f, Brow = 0.0f;
  const int myrow = r0 + tid;
  if (tid < RPB) {
    Brow = B[myrow];
    out[(size_t)myrow * TT]                      = 1.0f;  // s, column 0
    out[(size_t)NT     + (size_t)myrow * TT]     = 1.0f;  // g, column 0
    out[(size_t)2 * NT + (size_t)myrow * TT]     = 1.0f;  // x, column 0
  }

  // WMMA operand geometry (ISA 7.12.2, 16-bit A 16x32 / B 32x16):
  const int mrow  = lane & 15;               // A: M = lane mod 16
  const int khalf = (lane < 16) ? 0 : 8;     // A: K sub-half per lane group
  const int bbase = (lane < 16) ? 0 : 8;     // B: K sub-half per lane group
  const int kw    = wv * 256;                // this wave's K slice

  for (int t = 1; t < TT; ++t) {
    const __bf16* xc = (t & 1) ? xb0 : xb1;  // x_t broadcast (bf16)
    __bf16*       xn = (t & 1) ? xb1 : xb0;  // x_{t+1} broadcast

    v8f c = {};
#pragma unroll
    for (int i = 0; i < 8; ++i) {            // 8 x (16x16x32) = K span of 256
      const int kc = kw + i * 32;
      v8bf alo = *(const v8bf*)&Wt[mrow * NN + kc + khalf];        // K 0..7 | 8..15
      v8bf ahi = *(const v8bf*)&Wt[mrow * NN + kc + 16 + khalf];   // K 16..23 | 24..31
      v8bf blo = *(const v8bf*)&xc[kc + bbase];                    // K 0..7 | 8..15
      v8bf bhi = *(const v8bf*)&xc[kc + bbase + 16];               // K 16..23 | 24..31
      v16bf a, b;
#pragma unroll
      for (int j = 0; j < 8; ++j) {
        a[j] = alo[j]; a[8 + j] = ahi[j];
        b[j] = blo[j]; b[8 + j] = bhi[j];
      }
      c = __builtin_amdgcn_wmma_f32_16x16x32_bf16(
              false, a, false, b, (short)0, c, false, false);
    }

    // Column 0 of D: lane 0 holds rows 0..7, lane 16 holds rows 8..15.
    if (lane == 0) {
#pragma unroll
      for (int j = 0; j < 8; ++j) red[wv][j] = c[j];
    } else if (lane == 16) {
#pragma unroll
      for (int j = 0; j < 8; ++j) red[wv][8 + j] = c[j];
    }
    __syncthreads();

    if (tid < RPB) {
      float y = Brow;
#pragma unroll
      for (int wq = 0; wq < NWAV; ++wq) y += red[wq][tid];
      float g = __expf(y) + 1.0f;                        // nl = exp(x) + 1
      unsigned rs = (unsigned)myrow * 2654435761u ^ (unsigned)t * 40503u ^ 0x9E3779B9u;
      float s_new = poisson_sample(rs, g * DTCONST);     // spk ~ Poisson(g*dt)
      float x_new = decay * xr + sr;                     // x_{t+1} = (1-dt/tau)x_t + s_t
      out[(size_t)myrow * TT + t]                  = s_new;
      out[(size_t)NT     + (size_t)myrow * TT + t] = g;
      out[(size_t)2 * NT + (size_t)myrow * TT + t] = x_new;
      xr = x_new;
      sr = s_new;
      xn[myrow] = (__bf16)x_new;                         // publish x_{t+1}
    }
    grid_barrier(cnt, gen, NBLK);                        // one device sync per step
  }
}

extern "C" void kernel_launch(void* const* d_in, const int* in_sizes, int n_in,
                              void* d_out, int out_size, void* d_ws, size_t ws_size,
                              hipStream_t stream) {
  const float* W   = (const float*)d_in[0];
  const float* B   = (const float*)d_in[1];
  const float* tau = (const float*)d_in[2];
  float* out = (float*)d_out;

  // workspace layout: [xb0: 2048 bf16][xb1: 2048 bf16][cnt u32][gen u32]
  __bf16* xb0 = (__bf16*)d_ws;
  __bf16* xb1 = xb0 + NN;
  unsigned* cnt = (unsigned*)(xb1 + NN);
  unsigned* gen = cnt + 1;

  glmrnn_init<<<(NN + 255) / 256, 256, 0, stream>>>(xb0, cnt, gen);
  glmrnn_persistent<<<NBLK, NTHR, 0, stream>>>(W, B, tau, out, xb0, xb1, cnt, gen);
}
